// SCNN_45707041964767
// MI455X (gfx1250) — matrix operand order repaired and checked by reading, not compiled, not run on hardware
//
#include <hip/hip_runtime.h>
#include <hip/hip_bf16.h>

typedef __attribute__((ext_vector_type(16))) __bf16 v16bf;
typedef __attribute__((ext_vector_type(8)))  __bf16 v8bf;
typedef __attribute__((ext_vector_type(8)))  float  v8f;

#define C_CH 64
#define KW 9
#define CT_STRIDE 72      // bf16 elems per carry row: 64 ch + 8 pad (conflict spread, 16B-aligned slots)
#define KCHUNKS 18        // GEMM K = 64*9 = 576 = 18 chunks of 32

// ---------------------------------------------------------------------------
// Weight prep: f32 w[64][64][9] -> bf16 WMMA A-fragments.
// A-frag layout: [mtile(4)][chunk(18)][lane(32)][elem(16)] bf16, 32B per lane.
// Per CDNA5 ISA (16-bit A 16x32): lane m = mt*16 + (lane&15),
//   K_off(e, lane) = e + (e>=8 ? 8 : 0) + (lane>=16 ? 8 : 0), K = 32*c + K_off.
// K ordering chosen as K = tap*64 + in_ch so B-fragments are contiguous loads.
// ---------------------------------------------------------------------------
__global__ void wprep_kernel(const float* __restrict__ w, __bf16* __restrict__ afrag) {
  int idx = blockIdx.x * blockDim.x + threadIdx.x;
  if (idx >= 4 * KCHUNKS * 32 * 16) return;
  int e    = idx & 15;
  int lane = (idx >> 4) & 31;
  int rest = idx >> 9;
  int c    = rest % KCHUNKS;
  int mt   = rest / KCHUNKS;
  int m = mt * 16 + (lane & 15);
  int K = 32 * c + e + ((e >= 8) ? 8 : 0) + ((lane >= 16) ? 8 : 0);
  int k = K >> 6;   // tap index
  int i = K & 63;   // input channel
  afrag[idx] = (__bf16)w[(m * C_CH + i) * KW + k];
}

// ---------------------------------------------------------------------------
// Directional scan kernel. One workgroup per batch image (grid = B).
// Scans over S (stride sS), conv of width KW=9/pad 4 along contiguous dim N.
// Carry kept in LDS as bf16, transposed [row n][ch], double-buffered.
// step: out[s] = x[s] + relu(W * im2col(carry) + bias); carry = out[s].
// ---------------------------------------------------------------------------
__global__ __launch_bounds__(512)
void scan_kernel(const float* __restrict__ xin, float* __restrict__ out,
                 const __bf16* __restrict__ afrag, const float* __restrict__ bias,
                 int S, int N, int fwd, size_t sS, size_t cS, size_t bS) {
  extern __shared__ __align__(16) char smem_raw[];
  const int rows = N + 8;
  __bf16* buf0 = (__bf16*)smem_raw;
  __bf16* buf1 = buf0 + (size_t)rows * CT_STRIDE;

  const int b    = blockIdx.x;
  const int tid  = threadIdx.x;
  const int lane = tid & 31;
  const int wave = tid >> 5;          // 16 waves
  const int mt   = wave & 3;          // M-tile (16 out channels)
  const int ng   = wave >> 2;         // N-tile group 0..3
  const int jcnt = (N >> 4) >> 2;     // N-tiles per wave
  const int l15  = lane & 15;
  const int hi   = lane >> 4;
  const int mbase = mt * 16 + hi * 8; // first of this lane's 8 C/D channels

  const float* xb = xin + (size_t)b * bS;
  float*       ob = out + (size_t)b * bS;

  // Hold this wave's A fragments (weights) in VGPRs: 18 x v16bf = 144 VGPRs.
  v16bf areg[KCHUNKS];
  const v16bf* af = (const v16bf*)afrag;
#pragma unroll
  for (int c = 0; c < KCHUNKS; ++c)
    areg[c] = af[(mt * KCHUNKS + c) * 32 + lane];

  float bv[8];
#pragma unroll
  for (int r = 0; r < 8; ++r) bv[r] = bias[mbase + r];

  // Zero the 4+4 halo rows of both carry buffers (conv zero-padding).
  for (int t = tid; t < 2 * 8 * CT_STRIDE; t += blockDim.x) {
    int bb  = t / (8 * CT_STRIDE);
    int rem = t - bb * 8 * CT_STRIDE;
    int rr  = rem / CT_STRIDE;
    int cc  = rem - rr * CT_STRIDE;
    int row = (rr < 4) ? rr : (N + rr);
    (bb ? buf1 : buf0)[(size_t)row * CT_STRIDE + cc] = (__bf16)0.0f;
  }

  // Init: first row passes through unchanged; carry = that row (bf16, transposed).
  const int s0 = fwd ? 0 : (S - 1);
  for (int t = tid; t < C_CH * N; t += blockDim.x) {
    int ch = t / N;
    int n  = t - ch * N;
    float v = xb[(size_t)ch * cS + (size_t)s0 * sS + n];
    ob[(size_t)ch * cS + (size_t)s0 * sS + n] = v;
    buf0[(size_t)(n + 4) * CT_STRIDE + ch] = (__bf16)v;
  }
  __syncthreads();

  for (int h = 1; h < S; ++h) {
    const int s = fwd ? h : (S - 1 - h);
    const __bf16* bufR = (h & 1) ? buf0 : buf1;
    __bf16*       bufW = (h & 1) ? buf1 : buf0;

    for (int j = 0; j < jcnt; ++j) {
      const int nt = ng + 4 * j;
      const int n  = nt * 16 + l15;   // this lane's B column / D column
      v8f acc = {};
#pragma unroll
      for (int c = 0; c < KCHUNKS; ++c) {
        // B 32x16 bf16 (ISA layout): lane col n = lane&15, K_off = e + (e>=8?8:0) + (hi?8:0).
        // K = 32c + K_off -> tap k = c>>1, in_ch i = (c&1)*32 + K_off.
        // im2col[K][n] = carry[i][n + k - 4]; pad-4 storage => LDS row n + k.
        const int k  = c >> 1;
        const int i0 = ((c & 1) << 5) + (hi << 3);
        const __bf16* p = bufR + (size_t)(n + k) * CT_STRIDE + i0;
        union { v16bf v; v8bf h2[2]; } bf;
        bf.h2[0] = *(const v8bf*)(p);       // ds_load_b128
        bf.h2[1] = *(const v8bf*)(p + 16);  // ds_load_b128
        acc = __builtin_amdgcn_wmma_f32_16x16x32_bf16(
            false, areg[c], false, bf.v, (short)0, acc, false, false);
      }
      // Epilogue: bias + relu + residual; f32 to global, bf16 carry to LDS.
      v8bf cv;
#pragma unroll
      for (int r = 0; r < 8; ++r) {
        float conv = acc[r] + bv[r];
        conv = conv > 0.f ? conv : 0.f;
        size_t off = (size_t)(mbase + r) * cS + (size_t)s * sS + n;
        float o = xb[off] + conv;           // xb may alias ob (in-place passes): RAW same thread, safe
        ob[off] = o;
        cv[r] = (__bf16)o;
      }
      *(v8bf*)(bufW + (size_t)(n + 4) * CT_STRIDE + mbase) = cv;  // ds_store_b128
    }
    __syncthreads();  // all writes of step h visible before step h+1 reads
  }
}

// ---------------------------------------------------------------------------
// Plane transpose: in [Z][H][W] -> out [Z][W][H]. H, W multiples of 32.
// ---------------------------------------------------------------------------
__global__ void transpose_kernel(const float* __restrict__ in, float* __restrict__ out,
                                 int H, int W) {
  __shared__ float tile[32][33];
  const size_t plane = (size_t)H * W;
  const float* ip = in + (size_t)blockIdx.z * plane;
  float*       op = out + (size_t)blockIdx.z * plane;
  int h0 = blockIdx.y * 32, w0 = blockIdx.x * 32;
  int tx = threadIdx.x, ty = threadIdx.y;
  for (int r = ty; r < 32; r += 8)
    tile[r][tx] = ip[(size_t)(h0 + r) * W + (w0 + tx)];
  __syncthreads();
  for (int r = ty; r < 32; r += 8)
    op[(size_t)(w0 + r) * H + (h0 + tx)] = tile[tx][r];
}

extern "C" void kernel_launch(void* const* d_in, const int* in_sizes, int n_in,
                              void* d_out, int out_size, void* d_ws, size_t ws_size,
                              hipStream_t stream) {
  const float* x    = (const float*)d_in[0];
  const float* w_ud = (const float*)d_in[1];
  const float* b_ud = (const float*)d_in[2];
  const float* w_du = (const float*)d_in[3];
  const float* b_du = (const float*)d_in[4];
  const float* w_lr = (const float*)d_in[5];
  const float* b_lr = (const float*)d_in[6];
  const float* w_rl = (const float*)d_in[7];
  const float* b_rl = (const float*)d_in[8];
  float* y = (float*)d_out;

  const int B = 8, H = 256, W = 512;
  const size_t afrag_elems = (size_t)4 * KCHUNKS * 32 * 16;  // per direction

  __bf16* A_ud = (__bf16*)d_ws;
  __bf16* A_du = A_ud + afrag_elems;
  __bf16* A_lr = A_du + afrag_elems;
  __bf16* A_rl = A_lr + afrag_elems;
  float* t = (float*)((char*)d_ws +
                      ((4 * afrag_elems * sizeof(__bf16) + 255) & ~(size_t)255));

  (void)hipFuncSetAttribute((const void*)scan_kernel,
                            hipFuncAttributeMaxDynamicSharedMemorySize, 160000);

  const int wgrid = (int)((afrag_elems + 255) / 256);
  wprep_kernel<<<wgrid, 256, 0, stream>>>(w_ud, A_ud);
  wprep_kernel<<<wgrid, 256, 0, stream>>>(w_du, A_du);
  wprep_kernel<<<wgrid, 256, 0, stream>>>(w_lr, A_lr);
  wprep_kernel<<<wgrid, 256, 0, stream>>>(w_rl, A_rl);

  const size_t cS = (size_t)H * W;            // channel stride (same for both layouts)
  const size_t bS = (size_t)C_CH * H * W;     // batch stride
  const size_t shm_hw = (size_t)2 * (W + 8) * CT_STRIDE * sizeof(__bf16);  // ~150 KB
  const size_t shm_wh = (size_t)2 * (H + 8) * CT_STRIDE * sizeof(__bf16);  // ~76 KB

  // Pass 1/2: scan over H, conv along contiguous W, layout [B,C,H,W].
  scan_kernel<<<B, 512, shm_hw, stream>>>(x, y, A_ud, b_ud, H, W, 1, (size_t)W, cS, bS);
  scan_kernel<<<B, 512, shm_hw, stream>>>(y, y, A_du, b_du, H, W, 0, (size_t)W, cS, bS);

  // Transpose planes to [B,C,W,H] so the H-conv becomes contiguous.
  transpose_kernel<<<dim3(W / 32, H / 32, B * C_CH), dim3(32, 8), 0, stream>>>(y, t, H, W);

  // Pass 3/4: scan over W, conv along contiguous H.
  scan_kernel<<<B, 512, shm_wh, stream>>>(t, t, A_lr, b_lr, W, H, 1, (size_t)H, cS, bS);
  scan_kernel<<<B, 512, shm_wh, stream>>>(t, t, A_rl, b_rl, W, H, 0, (size_t)H, cS, bS);

  // Transpose back to [B,C,H,W] into d_out.
  transpose_kernel<<<dim3(H / 32, W / 32, B * C_CH), dim3(32, 8), 0, stream>>>(t, y, W, H);
}
